// TaskTaskGAT1Layer_60318520705359
// MI455X (gfx1250) — matrix operand order repaired and checked by reading, not compiled, not run on hardware
//
#include <hip/hip_runtime.h>
#include <cmath>

// ---------------------------------------------------------------------------
// Types for CDNA5 WMMA (wave32)
// ---------------------------------------------------------------------------
typedef __attribute__((ext_vector_type(16))) __bf16 v16bf;
typedef __attribute__((ext_vector_type(8)))  __bf16 v8bf;
typedef __attribute__((ext_vector_type(8)))  float  v8f;

#define HH 4      // heads
#define CC 32     // channels per head
#define INDIM 64  // input dim
#define WCOLS 288 // 128 (Wl) + 128 (Wr) + 32 (Wres) fused GEMM width
#define SA_STR 72 // padded LDS stride (bf16 elems) for A rows   (144 B, 16B aligned)
#define SB_STR 72 // padded LDS stride (bf16 elems) for B columns (144 B, 16B aligned)

// ---------------------------------------------------------------------------
// Small helpers
// ---------------------------------------------------------------------------
__device__ __forceinline__ float warp_sum32(float v) {
  #pragma unroll
  for (int o = 16; o > 0; o >>= 1) v += __shfl_xor(v, o, 32);
  return v;
}

__device__ __forceinline__ void atomicMaxF(float* addr, float v) {
  if (v >= 0.0f) atomicMax((int*)addr, __float_as_int(v));
  else           atomicMin((unsigned int*)addr, __float_as_uint(v));
}

// ---------------------------------------------------------------------------
// Fill a float buffer with a constant
// ---------------------------------------------------------------------------
__global__ void fill_kernel(float* __restrict__ p, float v, int n) {
  int i = blockIdx.x * blockDim.x + threadIdx.x;
  if (i < n) p[i] = v;
}

// ---------------------------------------------------------------------------
// x (f32, N x 64) -> bf16 copy for WMMA, and write skip-connection slice of out
// grid = N blocks of 64 threads
// ---------------------------------------------------------------------------
__global__ void prep_x_kernel(const float* __restrict__ x, __bf16* __restrict__ xbf,
                              float* __restrict__ out, int N_) {
  int n = blockIdx.x;
  int j = threadIdx.x;
  if (n >= N_) return;
  float v = x[(size_t)n * INDIM + j];
  xbf[(size_t)n * INDIM + j] = (__bf16)v;
  out[(size_t)n * 128 + 64 + j] = v;   // out = [dep(32) | dpt(32) | x(64)]
}

// ---------------------------------------------------------------------------
// Pack [Wl | Wr | Wres] into a bf16 64x288 matrix and [bl | br | bias] into bcat
// ---------------------------------------------------------------------------
__global__ void prep_w_kernel(const float* __restrict__ Wl, const float* __restrict__ Wr,
                              const float* __restrict__ Wres, const float* __restrict__ bl,
                              const float* __restrict__ br, const float* __restrict__ bias,
                              __bf16* __restrict__ wcat, float* __restrict__ bcat) {
  int t = blockIdx.x * blockDim.x + threadIdx.x;
  if (t < INDIM * WCOLS) {
    int k = t / WCOLS, j = t % WCOLS;
    float v;
    if (j < 128)      v = Wl[k * 128 + j];
    else if (j < 256) v = Wr[k * 128 + (j - 128)];
    else              v = Wres[k * 32 + (j - 256)];
    wcat[t] = (__bf16)v;
  }
  if (t < WCOLS) {
    bcat[t] = (t < 128) ? bl[t] : ((t < 256) ? br[t - 128] : bias[t - 256]);
  }
}

// ---------------------------------------------------------------------------
// Fused GEMM:  XW[N x 288] = x_bf16[N x 64] @ wcat_bf16[64 x 288] + bcat
// Block = 256 threads = 8 waves. blockIdx.x = 16-row tile, each wave = one
// 16x16 output tile via 2x v_wmma_f32_16x16x32_bf16.
// A rows staged in LDS row-major (padded), weights staged TRANSPOSED
// (column-major, padded) so every WMMA operand fragment is contiguous
// 16B runs -> ds_load_b128 instead of scalar u16 loads.
// ---------------------------------------------------------------------------
__global__ void gemm288_kernel(const __bf16* __restrict__ xbf,
                               const __bf16* __restrict__ wcat,
                               const float* __restrict__ bcat,
                               float* __restrict__ XW, int N_) {
  __shared__ __bf16 sA[16 * SA_STR];        // ~2.3 KB
  __shared__ __bf16 sBt[WCOLS * SB_STR];    // ~41.5 KB (transposed: [col][k])

  const int rowBase = blockIdx.x * 16;
  const int tid = threadIdx.x;

  // Stage A rows (zero-pad past N)
  for (int idx = tid; idx < 16 * INDIM; idx += blockDim.x) {
    int r = idx / INDIM, k = idx % INDIM;
    int gr = rowBase + r;
    sA[r * SA_STR + k] = (gr < N_) ? xbf[(size_t)gr * INDIM + k] : (__bf16)0.0f;
  }
  // Stage weights transposed: sBt[col][k] = wcat[k][col]
  for (int idx = tid; idx < INDIM * WCOLS; idx += blockDim.x) {
    int k = idx / WCOLS, c = idx % WCOLS;
    sBt[c * SB_STR + k] = wcat[idx];
  }
  __syncthreads();

  const int wave = tid >> 5;
  const int lane = tid & 31;
  const int colTile = blockIdx.y * 8 + wave;        // 18 column tiles of 16
  if (colTile >= WCOLS / 16) return;
  const int nBase = colTile * 16;

  const int half = lane >> 4;   // 0: lanes 0-15, 1: lanes 16-31
  const int mrow = lane & 15;

  v8f acc = {};
  #pragma unroll
  for (int ks = 0; ks < 2; ++ks) {          // K = 64 -> two K=32 WMMA steps
    const int kbase = ks * 32;
    // A (16x32 bf16) lane layout: lanes 0-15 hold K {0..7,16..23},
    // lanes 16-31 hold K {8..15,24..31} -> two contiguous 8-elt runs.
    const __bf16* pa = &sA[mrow * SA_STR + kbase + 8 * half];
    v8bf a0 = *(const v8bf*)(pa);
    v8bf a1 = *(const v8bf*)(pa + 16);
    v16bf a = __builtin_shufflevector(a0, a1, 0, 1, 2, 3, 4, 5, 6, 7,
                                              8, 9, 10, 11, 12, 13, 14, 15);
    // B (32x16 bf16) lane layout: lanes 0-15 hold K {0..7,16..23},
    // lanes 16-31 hold K {8..15,24..31}, col = lane%16 -> two contiguous 8-elt runs.
    const __bf16* pb = &sBt[(nBase + mrow) * SB_STR + kbase + 8 * half];
    v8bf b0 = *(const v8bf*)(pb);
    v8bf b1 = *(const v8bf*)(pb + 16);
    v16bf b = __builtin_shufflevector(b0, b1, 0, 1, 2, 3, 4, 5, 6, 7,
                                              8, 9, 10, 11, 12, 13, 14, 15);
    acc = __builtin_amdgcn_wmma_f32_16x16x32_bf16(
        /*neg_a=*/false, a, /*neg_b=*/false, b,
        /*c_mod=*/(short)0, acc, /*reuse_a=*/false, /*reuse_b=*/false);
  }

  // C/D layout: lanes 0-15 -> rows 0..7 (vgpr i), lanes 16-31 -> rows 8..15
  const int col = nBase + mrow;
  const float bv = bcat[col];
  float* outp = XW + (size_t)(rowBase + 8 * half) * WCOLS + col;
  if (rowBase + 16 <= N_) {
    // full tile: wave-uniform unguarded stores
    #pragma unroll
    for (int i = 0; i < 8; ++i) outp[(size_t)i * WCOLS] = acc[i] + bv;
  } else {
    #pragma unroll
    for (int i = 0; i < 8; ++i) {
      if (rowBase + 8 * half + i < N_) outp[(size_t)i * WCOLS] = acc[i] + bv;
    }
  }
}

// ---------------------------------------------------------------------------
// Edge attention score + segment max.  One wave per edge; lane = channel.
// score[e,h] = sum_c att[h,c] * leaky_relu(xl[src,h,c] + xr[dst,h,c], 0.2)
// ---------------------------------------------------------------------------
__global__ void edge_score_kernel(const int* __restrict__ srcI, const int* __restrict__ dstI,
                                  const float* __restrict__ XW, const float* __restrict__ att,
                                  float* __restrict__ score, float* __restrict__ mmax, int E_) {
  int e = blockIdx.x * (blockDim.x >> 5) + (threadIdx.x >> 5);
  int lane = threadIdx.x & 31;
  if (e >= E_) return;
  int dN = dstI[e];
  size_t sb = (size_t)srcI[e] * WCOLS;        // xl slice: cols 0..127
  size_t db = (size_t)dN * WCOLS + 128;       // xr slice: cols 128..255
  float sc[HH];
  #pragma unroll
  for (int h = 0; h < HH; ++h) {
    float v = XW[sb + h * CC + lane] + XW[db + h * CC + lane];
    v = (v > 0.0f) ? v : 0.2f * v;
    sc[h] = warp_sum32(att[h * CC + lane] * v);
  }
  if (lane < HH) {
    float p = sc[lane];
    score[(size_t)e * HH + lane] = p;
    atomicMaxF(&mmax[(size_t)dN * HH + lane], p);
  }
}

// ---------------------------------------------------------------------------
// ex = exp(score - max[dst]); accumulate denominator. Thread per (edge, head).
// ---------------------------------------------------------------------------
__global__ void edge_exp_kernel(const int* __restrict__ dstI, const float* __restrict__ mmax,
                                float* __restrict__ score, float* __restrict__ den, int EH) {
  int i = blockIdx.x * blockDim.x + threadIdx.x;
  if (i >= EH) return;
  int e = i >> 2, h = i & 3;
  size_t di = (size_t)dstI[e] * HH + h;
  float ex = __expf(score[i] - mmax[di]);
  score[i] = ex;
  atomicAdd(&den[di], ex);
}

// ---------------------------------------------------------------------------
// acc[dst,h,c] += xl[src,h,c] * (ex / (den[dst,h] + 1e-16)).  Wave per edge.
// ---------------------------------------------------------------------------
__global__ void edge_scatter_kernel(const int* __restrict__ srcI, const int* __restrict__ dstI,
                                    const float* __restrict__ XW, const float* __restrict__ score,
                                    const float* __restrict__ den, float* __restrict__ accum, int E_) {
  int e = blockIdx.x * (blockDim.x >> 5) + (threadIdx.x >> 5);
  int lane = threadIdx.x & 31;
  if (e >= E_) return;
  int sN = srcI[e], dN = dstI[e];
  size_t sb = (size_t)sN * WCOLS;
  size_t ab = (size_t)dN * (HH * CC);
  #pragma unroll
  for (int h = 0; h < HH; ++h) {
    float alpha = score[(size_t)e * HH + h] / (den[(size_t)dN * HH + h] + 1e-16f);
    atomicAdd(&accum[ab + h * CC + lane], XW[sb + h * CC + lane] * alpha);
  }
}

// ---------------------------------------------------------------------------
// Finalize: mean over heads + residual (XW cols 256..287 already has bias),
// LayerNorm over 32 channels, leaky_relu(0.01), write to out column slice.
// Wave per node, lane = channel.
// ---------------------------------------------------------------------------
__global__ void finalize_kernel(const float* __restrict__ accum, const float* __restrict__ XW,
                                const float* __restrict__ g, const float* __restrict__ b,
                                float* __restrict__ out, int N_, int colOff) {
  int n = blockIdx.x * (blockDim.x >> 5) + (threadIdx.x >> 5);
  int lane = threadIdx.x & 31;
  if (n >= N_) return;
  size_t ab = (size_t)n * (HH * CC);
  float v = 0.25f * (accum[ab + lane] + accum[ab + 32 + lane] +
                     accum[ab + 64 + lane] + accum[ab + 96 + lane])
          + XW[(size_t)n * WCOLS + 256 + lane];
  float mu  = warp_sum32(v) * (1.0f / 32.0f);
  float d   = v - mu;
  float var = warp_sum32(d * d) * (1.0f / 32.0f);
  float y = d * rsqrtf(var + 1e-5f) * g[lane] + b[lane];
  y = (y > 0.0f) ? y : 0.01f * y;
  out[(size_t)n * 128 + colOff + lane] = y;
}

// ---------------------------------------------------------------------------
// Launch
// ---------------------------------------------------------------------------
extern "C" void kernel_launch(void* const* d_in, const int* in_sizes, int n_in,
                              void* d_out, int out_size, void* d_ws, size_t ws_size,
                              hipStream_t stream) {
  const float* x  = (const float*)d_in[0];
  const int*   ei = (const int*)d_in[1];
  const int N_ = in_sizes[0] / INDIM;
  const int E_ = in_sizes[1] / 2;
  const int* srcAll = ei;
  const int* dstAll = ei + E_;

  // per-conv parameter pointers: Wl, bl, Wr, br, att, Wres, bias at d_in[2+7c .. 8+7c]
  const float* Wl[2]   = {(const float*)d_in[2],  (const float*)d_in[9]};
  const float* bl[2]   = {(const float*)d_in[3],  (const float*)d_in[10]};
  const float* Wr[2]   = {(const float*)d_in[4],  (const float*)d_in[11]};
  const float* br[2]   = {(const float*)d_in[5],  (const float*)d_in[12]};
  const float* att[2]  = {(const float*)d_in[6],  (const float*)d_in[13]};
  const float* Wrs[2]  = {(const float*)d_in[7],  (const float*)d_in[14]};
  const float* bia[2]  = {(const float*)d_in[8],  (const float*)d_in[15]};
  const float* lng[2]  = {(const float*)d_in[16], (const float*)d_in[18]};
  const float* lnb[2]  = {(const float*)d_in[17], (const float*)d_in[19]};

  float* out = (float*)d_out;

  // Workspace carve-up (256B aligned)
  char* ws = (char*)d_ws;
  size_t off = 0;
  auto carve = [&](size_t bytes) -> void* {
    void* p = ws + off;
    off = (off + bytes + 255) & ~(size_t)255;
    return p;
  };
  __bf16* xbf   = (__bf16*)carve((size_t)N_ * INDIM * sizeof(__bf16));
  __bf16* wcat  = (__bf16*)carve((size_t)2 * INDIM * WCOLS * sizeof(__bf16));
  float*  bcat  = (float*) carve((size_t)2 * WCOLS * sizeof(float));
  float*  XW    = (float*) carve((size_t)N_ * WCOLS * sizeof(float));   // reused per conv
  float*  score = (float*) carve((size_t)E_ * HH * sizeof(float));
  float*  mmax  = (float*) carve((size_t)N_ * HH * sizeof(float));
  float*  den   = (float*) carve((size_t)N_ * HH * sizeof(float));
  float*  accum = (float*) carve((size_t)N_ * HH * CC * sizeof(float));

  // Prep: bf16 input + skip-connection copy; weight packing
  prep_x_kernel<<<N_, INDIM, 0, stream>>>(x, xbf, out, N_);
  for (int c = 0; c < 2; ++c) {
    prep_w_kernel<<<(INDIM * WCOLS + 255) / 256, 256, 0, stream>>>(
        Wl[c], Wr[c], Wrs[c], bl[c], br[c], bia[c],
        wcat + (size_t)c * INDIM * WCOLS, bcat + (size_t)c * WCOLS);
  }

  const int rowTiles = (N_ + 15) / 16;
  const int EH = E_ * HH;

  for (int c = 0; c < 2; ++c) {
    // init accumulators / softmax stats
    fill_kernel<<<((N_ * HH * CC) + 255) / 256, 256, 0, stream>>>(accum, 0.0f, N_ * HH * CC);
    fill_kernel<<<((N_ * HH) + 255) / 256, 256, 0, stream>>>(mmax, -INFINITY, N_ * HH);
    fill_kernel<<<((N_ * HH) + 255) / 256, 256, 0, stream>>>(den, 0.0f, N_ * HH);

    // fused WMMA GEMM -> XW = [xl | xr | res+bias]
    dim3 ggrid(rowTiles, 3);  // 8 waves/block x 3 = 24 >= 18 col tiles
    gemm288_kernel<<<ggrid, 256, 0, stream>>>(
        xbf, wcat + (size_t)c * INDIM * WCOLS, bcat + (size_t)c * WCOLS, XW, N_);

    // dep uses (src -> dst); dpt flips the edges
    const int* s = (c == 0) ? srcAll : dstAll;
    const int* d = (c == 0) ? dstAll : srcAll;

    edge_score_kernel<<<(E_ + 7) / 8, 256, 0, stream>>>(s, d, XW, att[c], score, mmax, E_);
    edge_exp_kernel<<<(EH + 255) / 256, 256, 0, stream>>>(d, mmax, score, den, EH);
    edge_scatter_kernel<<<(E_ + 7) / 8, 256, 0, stream>>>(s, d, XW, score, den, accum, E_);
    finalize_kernel<<<(N_ + 7) / 8, 256, 0, stream>>>(accum, XW, lng[c], lnb[c], out, N_, c * CC);
  }
}
